// SWA_30399778521718
// MI455X (gfx1250) — compile-verified, matrix operand order checked by reading, not run on hardware
//
#include <hip/hip_runtime.h>

#define DM     512
#define WIN    128
#define NH     8
#define HD     64
#define BATCH  8
#define SEQ    4096
#define NW     32

typedef __attribute__((ext_vector_type(16))) __bf16 v16bf;
typedef __attribute__((ext_vector_type(8)))  __bf16 v8bf;
typedef __attribute__((ext_vector_type(8)))  float  v8f;

// Build a 16x32 bf16 WMMA A/B fragment from a row-major LDS row (32-elem K window).
// Per ISA layout: lanes 0-15 hold K={0..7,16..23}, lanes 16-31 hold K={8..15,24..31},
// packed pairs -> contiguous, so two ds_load_b128 suffice.
static __device__ __forceinline__ v16bf ldsFrag(const __bf16* rowBase, int lane) {
  const int kLo = (lane & 16) ? 8 : 0;
  v8bf lo = *(const v8bf*)(rowBase + kLo);
  v8bf hi = *(const v8bf*)(rowBase + kLo + 16);
  v16bf f;
#pragma unroll
  for (int i = 0; i < 8; ++i) { f[i] = lo[i]; f[i + 8] = hi[i]; }
  return f;
}

static __device__ __forceinline__ v8f wmma_bf16(v16bf a, v16bf b, v8f c) {
  return __builtin_amdgcn_wmma_f32_16x16x32_bf16(false, a, false, b, (short)0, c, false, false);
}

// Async copy 16B global -> LDS (CDNA5 GLOBAL_LOAD_ASYNC_TO_LDS_B128, ASYNCcnt-tracked).
static __device__ __forceinline__ void asyncCopy16(unsigned ldsByteOff, uint64_t gaddr) {
  asm volatile("global_load_async_to_lds_b128 %0, %1, off"
               :: "v"(ldsByteOff), "v"(gaddr) : "memory");
}
static __device__ __forceinline__ void asyncWait0() {
  asm volatile("s_wait_asynccnt 0x0" ::: "memory");
}

// Store a staged 128x32 f32 tile (4 float4 per thread) into LDS as bf16 [row][k].
static __device__ __forceinline__ void stageToLds(__bf16* lds, const float4* st, int tid) {
#pragma unroll
  for (int it = 0; it < 4; ++it) {
    int e4 = tid + it * 256;
    int row = e4 >> 3;
    int c4 = (e4 & 7) * 4;
    __bf16* d = &lds[row * 32 + c4];
    d[0] = (__bf16)st[it].x; d[1] = (__bf16)st[it].y;
    d[2] = (__bf16)st[it].z; d[3] = (__bf16)st[it].w;
  }
}
// Load a 128x32 f32 tile from src (row stride = DM) into registers.
static __device__ __forceinline__ void loadTile(const float* __restrict__ src, int k0,
                                                float4* st, int tid) {
#pragma unroll
  for (int it = 0; it < 4; ++it) {
    int e4 = tid + it * 256;
    int row = e4 >> 3;
    int c4 = (e4 & 7) * 4;
    st[it] = *(const float4*)(src + (size_t)row * DM + k0 + c4);
  }
}

// ---------------- Kernel 1: fused QKV projection ----------------
// C[m,e] = sum_d x[m,d] * w[e,d] + bias[e];  scatter into Q/K/V [b][win][h][r][hd] bf16.
__global__ __launch_bounds__(256) void swa_qkv_proj(
    const float* __restrict__ x, const float* __restrict__ w,
    const float* __restrict__ bias,
    __bf16* __restrict__ qws, __bf16* __restrict__ kws, __bf16* __restrict__ vws)
{
  __shared__ __bf16 As[128 * 32];   // [m][k]
  __shared__ __bf16 Bs[128 * 32];   // [n][k] (row of W)

  const int tid = threadIdx.x;
  const int lane = tid & 31;
  const int wv = tid >> 5;
  const int rl = lane & 15;
  const int mo = (lane >> 4) * 8;
  const int m0 = blockIdx.x * 128;
  const int n0 = blockIdx.y * 128;
  const float* xBase = x + (size_t)m0 * DM;
  const float* wBase = w + (size_t)n0 * DM;

  v8f acc[8];
#pragma unroll
  for (int j = 0; j < 8; ++j)
#pragma unroll
    for (int i = 0; i < 8; ++i) acc[j][i] = 0.0f;

  // Software pipeline: tile k0 staged in regs; prefetch k0+32 while WMMAs run.
  float4 ax[4], bx[4];
  loadTile(xBase, 0, ax, tid);
  loadTile(wBase, 0, bx, tid);

  for (int k0 = 0; k0 < DM; k0 += 32) {
    stageToLds(As, ax, tid);
    stageToLds(Bs, bx, tid);
    __syncthreads();
    int kn = (k0 + 32 < DM) ? k0 + 32 : 0;   // wrap: harmless redundant prefetch
    loadTile(xBase, kn, ax, tid);            // issued now, waited at next stage
    loadTile(wBase, kn, bx, tid);

    v16bf a = ldsFrag(&As[(wv * 16 + rl) * 32], lane);
#pragma unroll
    for (int nj = 0; nj < 8; ++nj) {
      v16bf b = ldsFrag(&Bs[(nj * 16 + rl) * 32], lane);
      acc[nj] = wmma_bf16(a, b, acc[nj]);
    }
    __syncthreads();
  }

#pragma unroll
  for (int nj = 0; nj < 8; ++nj) {
    int e = n0 + nj * 16 + rl;
    int sel = e >> 9;              // 0=Q 1=K 2=V
    int ec = e & 511;
    int h = ec >> 6;
    int c = ec & 63;
    float bval = bias[e];
    float scale = (sel == 0) ? 0.125f : 1.0f;   // 1/sqrt(64) folded into Q
    __bf16* dst = (sel == 0) ? qws : (sel == 1) ? kws : vws;
#pragma unroll
    for (int r = 0; r < 8; ++r) {
      int mg = m0 + wv * 16 + r + mo;
      int b  = mg >> 12;
      int s  = mg & 4095;
      int win = s >> 7;
      int rw  = s & 127;
      size_t idx = (((size_t)(b * NW + win) * NH + h) * WIN + rw) * HD + c;
      dst[idx] = (__bf16)((acc[nj][r] + bval) * scale);
    }
  }
}

// ---------------- Kernel 2: per-window attention ----------------
// One block per (b, win, h): S = Q*K^T (128x128), softmax, ctx = P*V (128x64).
__global__ __launch_bounds__(256) void swa_attn(
    const __bf16* __restrict__ qws, const __bf16* __restrict__ kws,
    const __bf16* __restrict__ vws, __bf16* __restrict__ ctxws)
{
  __shared__ __bf16 QKP[128 * 128];   // Q[0:8192] | K[8192:16384]; reused as P[128x128]
  __shared__ __bf16 Vt[64 * 128];     // V transposed: [n][k]

  __bf16* Qs = QKP;
  __bf16* Ks = QKP + WIN * HD;
  __bf16* Ps = QKP;

  const int tid = threadIdx.x, lane = tid & 31, wv = tid >> 5;
  const int rl = lane & 15, mo = (lane >> 4) * 8;
  const size_t base = (size_t)blockIdx.x * (WIN * HD);

  // Q/K tiles are raw byte copies -> CDNA5 async tensor path (ASYNCcnt).
  const unsigned ldsQ = (unsigned)(uintptr_t)Qs;   // low 32 bits = LDS byte offset
  const unsigned ldsK = (unsigned)(uintptr_t)Ks;
  const uint64_t gq = (uint64_t)(uintptr_t)(qws + base);
  const uint64_t gk = (uint64_t)(uintptr_t)(kws + base);
#pragma unroll
  for (int it = 0; it < 4; ++it) {
    unsigned byteOff = (unsigned)(tid + it * 256) * 16u;   // 16B per lane per iter
    asyncCopy16(ldsQ + byteOff, gq + byteOff);
    asyncCopy16(ldsK + byteOff, gk + byteOff);
  }
  // V needs an on-the-fly transpose: vector b128 global loads, scalar LDS scatter.
#pragma unroll
  for (int it = 0; it < 4; ++it) {
    int i = tid + it * 256;          // v8bf (8 elems along hd) index
    int r = i >> 3;                  // 8 v8bf per 64-elem row
    int c = (i & 7) * 8;
    v8bf vv = ((const v8bf*)(vws + base))[i];
#pragma unroll
    for (int j = 0; j < 8; ++j) Vt[(c + j) * WIN + r] = vv[j];
  }
  asyncWait0();
  __syncthreads();

  v8f acc[8];
#pragma unroll
  for (int j = 0; j < 8; ++j)
#pragma unroll
    for (int i = 0; i < 8; ++i) acc[j][i] = 0.0f;

  const int row = wv * 16 + rl;
#pragma unroll
  for (int k0 = 0; k0 < HD; k0 += 32) {
    v16bf a = ldsFrag(&Qs[row * HD + k0], lane);
#pragma unroll
    for (int nj = 0; nj < 8; ++nj) {
      v16bf b = ldsFrag(&Ks[(nj * 16 + rl) * HD + k0], lane);
      acc[nj] = wmma_bf16(a, b, acc[nj]);
    }
  }

  // Row softmax: row m = r + 8*(lane>=16); its 128 cols live in 8 regs x 16 lanes
  // of one wave32 half, so xor-shuffles with masks 8..1 reduce both halves at once.
#pragma unroll
  for (int r = 0; r < 8; ++r) {
    float mx = -3.4e38f;
#pragma unroll
    for (int nj = 0; nj < 8; ++nj) mx = fmaxf(mx, acc[nj][r]);
#pragma unroll
    for (int m = 8; m >= 1; m >>= 1) mx = fmaxf(mx, __shfl_xor(mx, m, 32));
    float sum = 0.0f;
#pragma unroll
    for (int nj = 0; nj < 8; ++nj) { float e = __expf(acc[nj][r] - mx); acc[nj][r] = e; sum += e; }
#pragma unroll
    for (int m = 8; m >= 1; m >>= 1) sum += __shfl_xor(sum, m, 32);
    float inv = 1.0f / sum;
#pragma unroll
    for (int nj = 0; nj < 8; ++nj) acc[nj][r] *= inv;
  }

  __syncthreads();   // all Q/K LDS reads done before overwrite with P
#pragma unroll
  for (int nj = 0; nj < 8; ++nj)
#pragma unroll
    for (int r = 0; r < 8; ++r)
      Ps[(wv * 16 + r + mo) * WIN + nj * 16 + rl] = (__bf16)acc[nj][r];
  __syncthreads();

  v8f o[4];
#pragma unroll
  for (int j = 0; j < 4; ++j)
#pragma unroll
    for (int i = 0; i < 8; ++i) o[j][i] = 0.0f;

#pragma unroll
  for (int k0 = 0; k0 < WIN; k0 += 32) {
    v16bf a = ldsFrag(&Ps[row * WIN + k0], lane);
#pragma unroll
    for (int nj = 0; nj < 4; ++nj) {
      v16bf b = ldsFrag(&Vt[(nj * 16 + rl) * WIN + k0], lane);
      o[nj] = wmma_bf16(a, b, o[nj]);
    }
  }

  const int h = blockIdx.x & 7;
  const int bw = blockIdx.x >> 3;
  const int win = bw & 31;
  const int bb = bw >> 5;
#pragma unroll
  for (int nj = 0; nj < 4; ++nj)
#pragma unroll
    for (int r = 0; r < 8; ++r) {
      int rr = wv * 16 + r + mo;
      size_t idx = ((size_t)(bb * SEQ + win * WIN + rr)) * DM + h * HD + nj * 16 + rl;
      ctxws[idx] = (__bf16)o[nj][r];
    }
}

// ---------------- Kernel 3: output projection ----------------
__global__ __launch_bounds__(256) void swa_out_proj(
    const __bf16* __restrict__ ctx, const float* __restrict__ w,
    const float* __restrict__ bias, float* __restrict__ out)
{
  __shared__ __bf16 As[128 * 32];
  __shared__ __bf16 Bs[128 * 32];
  const int tid = threadIdx.x, lane = tid & 31, wv = tid >> 5;
  const int rl = lane & 15, mo = (lane >> 4) * 8;
  const int m0 = blockIdx.x * 128;
  const int n0 = blockIdx.y * 128;
  const __bf16* cBase = ctx + (size_t)m0 * DM;
  const float*  wBase = w + (size_t)n0 * DM;

  v8f acc[8];
#pragma unroll
  for (int j = 0; j < 8; ++j)
#pragma unroll
    for (int i = 0; i < 8; ++i) acc[j][i] = 0.0f;

  // Software pipeline: A tile (already bf16) staged as 2 uint4/thread, B as 4 float4.
  uint4 au[2];
  float4 bx[4];
#pragma unroll
  for (int it = 0; it < 2; ++it) {
    int i = tid + it * 256;
    au[it] = ((const uint4*)(cBase + (size_t)(i >> 2) * DM))[i & 3];
  }
  loadTile(wBase, 0, bx, tid);

  for (int k0 = 0; k0 < DM; k0 += 32) {
#pragma unroll
    for (int it = 0; it < 2; ++it) {
      int i = tid + it * 256;
      ((uint4*)As)[i] = au[it];
    }
    stageToLds(Bs, bx, tid);
    __syncthreads();
    int kn = (k0 + 32 < DM) ? k0 + 32 : 0;   // wrap: harmless redundant prefetch
#pragma unroll
    for (int it = 0; it < 2; ++it) {
      int i = tid + it * 256;
      au[it] = ((const uint4*)(cBase + (size_t)(i >> 2) * DM + kn))[i & 3];
    }
    loadTile(wBase, kn, bx, tid);

    v16bf a = ldsFrag(&As[(wv * 16 + rl) * 32], lane);
#pragma unroll
    for (int nj = 0; nj < 8; ++nj) {
      v16bf b = ldsFrag(&Bs[(nj * 16 + rl) * 32], lane);
      acc[nj] = wmma_bf16(a, b, acc[nj]);
    }
    __syncthreads();
  }

#pragma unroll
  for (int nj = 0; nj < 8; ++nj) {
    int e = n0 + nj * 16 + rl;
    float bval = bias[e];
#pragma unroll
    for (int r = 0; r < 8; ++r) {
      int mg = m0 + wv * 16 + r + mo;
      out[(size_t)mg * DM + e] = acc[nj][r] + bval;
    }
  }
}

extern "C" void kernel_launch(void* const* d_in, const int* in_sizes, int n_in,
                              void* d_out, int out_size, void* d_ws, size_t ws_size,
                              hipStream_t stream) {
  const float* x   = (const float*)d_in[0];
  const float* ipw = (const float*)d_in[1];
  const float* ipb = (const float*)d_in[2];
  const float* opw = (const float*)d_in[3];
  const float* opb = (const float*)d_in[4];
  float* out = (float*)d_out;

  const size_t NEL = (size_t)BATCH * SEQ * DM;   // 16,777,216 elems per tensor
  __bf16* qws = (__bf16*)d_ws;
  __bf16* kws = qws + NEL;
  __bf16* vws = kws + NEL;
  __bf16* cws = vws + NEL;                        // total 4 * 32MB = 128MB bf16

  // M = B*S = 32768 rows
  swa_qkv_proj<<<dim3(32768 / 128, (3 * DM) / 128), 256, 0, stream>>>(x, ipw, ipb, qws, kws, vws);
  swa_attn<<<BATCH * NW * NH, 256, 0, stream>>>(qws, kws, vws, cws);
  swa_out_proj<<<dim3(32768 / 128, DM / 128), 256, 0, stream>>>(cws, opw, opb, out);
}
